// ParallelDroplessMLP_2353642078238
// MI455X (gfx1250) — matrix operand order repaired and checked by reading, not compiled, not run on hardware
//
#include <hip/hip_runtime.h>

// ---------------------------------------------------------------------------
// MoE dropless MLP for MI455X (gfx1250).
// Algebraic reduction: all experts share w1/w2 in the reference, so
//   out[t] = (ew[t,0]+ew[t,1]) * ( gelu(x[t] @ w1) @ w2 )
// 137 GFLOP, weights L2-resident (192 MB L2) -> compute-bound -> bf16 WMMA
// (v_wmma_f32_16x16x32_bf16, fp32 accumulate).
//
// v2: X tile staged once to LDS via GLOBAL_LOAD_ASYNC_TO_LDS_B128 (ASYNCcnt),
//     ownership re-partitioned so every w1/w2 fragment is loaded by at most
//     1-2 waves and reused in registers: ~12 GB -> ~3 GB L2 traffic.
// ---------------------------------------------------------------------------

#define T_TOKENS 4096
#define HID      1024
#define INTER    8192
#define BM       64      // tokens per workgroup
#define BK2      64      // inter-dim chunk per fused iteration
#define XS_STRIDE 1032   // elems: 2064 B rows -> 16B aligned, bank-conflict-free
#define HS_STRIDE 72     // elems:  144 B rows -> 16B aligned, bank-conflict-free

typedef __attribute__((ext_vector_type(16))) __bf16 bf16x16;
typedef __attribute__((ext_vector_type(8)))  float  f32x8;

union ABfrag { bf16x16 v; uint4 q[2]; };

__device__ __forceinline__ unsigned short f32_to_bf16(float f) {
    unsigned int u = __float_as_uint(f);
    u += 0x7FFFu + ((u >> 16) & 1u);     // round-to-nearest-even
    return (unsigned short)(u >> 16);
}

__device__ __forceinline__ float gelu_tanh(float v) {
    const float k = 0.7978845608028654f; // sqrt(2/pi)
    float u = k * (v + 0.044715f * v * v * v);
    return 0.5f * v * (1.0f + tanhf(u));
}

#define WMMA_BF16(A, B, C) \
    __builtin_amdgcn_wmma_f32_16x16x32_bf16(false, (A), false, (B), (short)0, (C), false, false)

// ---------------------------- prep kernels ---------------------------------

__global__ void k_cvt_x(const float* __restrict__ x, unsigned short* __restrict__ xb) {
    int i = blockIdx.x * 256 + threadIdx.x;          // exact: T*HID
    xb[i] = f32_to_bf16(x[i]);
}

__global__ void k_wsum(const float* __restrict__ ew, float* __restrict__ wsum) {
    int t = blockIdx.x * 256 + threadIdx.x;
    if (t < T_TOKENS) wsum[t] = ew[2 * t] + ew[2 * t + 1];
}

// w1: [HID][INTER] row-major  ->  w1t: [INTER][HID] bf16
__global__ void k_tr_w1(const float* __restrict__ w1, unsigned short* __restrict__ w1t) {
    int i = blockIdx.x * 256 + threadIdx.x;          // exact: INTER*HID
    int n = i / HID, k = i % HID;
    w1t[i] = f32_to_bf16(w1[k * INTER + n]);
}

// w2: [INTER][HID] row-major  ->  w2t: [HID][INTER] bf16
__global__ void k_tr_w2(const float* __restrict__ w2, unsigned short* __restrict__ w2t) {
    int i = blockIdx.x * 256 + threadIdx.x;          // exact: HID*INTER
    int n = i / INTER, k = i % INTER;
    w2t[i] = f32_to_bf16(w2[k * HID + n]);
}

// ---------------------------- fused MLP ------------------------------------
// Grid: 64 blocks x 256 threads (8 wave32 waves). Per chunk c:
//   phase1: H[64][64] = gelu(Xs @ W1chunk) ; wave owns 1 N-strip x 2 M-tiles
//           (B fragment loaded once from global, reused x2; A from LDS Xs)
//   phase2: Y[64][1024] += H @ W2chunk ; wave owns 8 distinct N-tiles x 4 M
//           (each B fragment loaded by exactly one wave, reused x4)

__global__ __launch_bounds__(256)
void k_moe_mlp(const unsigned short* __restrict__ xb,
               const unsigned short* __restrict__ w1t,
               const unsigned short* __restrict__ w2t,
               const float* __restrict__ wsum,
               float* __restrict__ out) {
    __shared__ unsigned short Xs[BM * XS_STRIDE];    // 132,096 B
    __shared__ unsigned short Hs[BM * HS_STRIDE];    //   9,216 B

    const int tid    = threadIdx.x;
    const int lane   = tid & 31;
    const int wave   = tid >> 5;
    const int lanelo = lane & 15;
    const int lhalf  = lane >> 4;                    // 0 or 1
    const int rowbase = blockIdx.x * BM;

    // ---- stage X tile (64 rows x 1024 bf16) into LDS, async DMA path ----
    {
        const char* src = (const char*)(xb + rowbase * HID);
        const unsigned ldsbase = (unsigned)(unsigned long long)(const void*)Xs;
        #pragma unroll
        for (int it = 0; it < 32; ++it) {
            const int cid = it * 256 + tid;          // 8192 16-byte chunks
            const int row = cid >> 7;                // 0..63
            const int col = (cid & 127) * 16;        // byte offset within row
            const unsigned dst = ldsbase + (unsigned)(row * (XS_STRIDE * 2) + col);
            const unsigned long long ga =
                (unsigned long long)(src + row * (HID * 2) + col);
            asm volatile("global_load_async_to_lds_b128 %0, %1, off"
                         :: "v"(dst), "v"(ga) : "memory");
        }
        asm volatile("s_wait_asynccnt 0x0" ::: "memory");
        __syncthreads();
    }

    // phase-1 ownership: N strip nt1, M tiles {mta, mta+1}
    const int nt1 = wave >> 1;                       // 0..3
    const int mta = (wave & 1) * 2;                  // 0 or 2
    // phase-2 ownership: N tiles wave*8+j (j<8), all 4 M tiles
    f32x8 acc[4][8];
    #pragma unroll
    for (int mt = 0; mt < 4; ++mt)
        #pragma unroll
        for (int j = 0; j < 8; ++j) acc[mt][j] = (f32x8)(0.0f);

    for (int c = 0; c < INTER / BK2; ++c) {
        const int ci = c * BK2;

        // -------- phase 1 --------
        // B lane layout (32x16): half-dependent K sub-blocks, same as A
        const unsigned short* bp1 = w1t + (ci + nt1 * 16 + lanelo) * HID + lhalf * 8;
        // A lane layout (16-bit 16x32): half-dependent K sub-blocks
        const int am0 = (mta * 16 + lanelo) * XS_STRIDE + lhalf * 8;
        const int am1 = am0 + 16 * XS_STRIDE;

        f32x8 h0 = (f32x8)(0.0f), h1 = (f32x8)(0.0f);
        #pragma unroll 4
        for (int k0 = 0; k0 < HID; k0 += 32) {
            ABfrag b, a0, a1;
            b.q[0]  = *(const uint4*)(bp1 + k0);
            b.q[1]  = *(const uint4*)(bp1 + k0 + 16);
            a0.q[0] = *(const uint4*)(&Xs[am0 + k0]);
            a0.q[1] = *(const uint4*)(&Xs[am0 + k0 + 16]);
            a1.q[0] = *(const uint4*)(&Xs[am1 + k0]);
            a1.q[1] = *(const uint4*)(&Xs[am1 + k0 + 16]);
            h0 = WMMA_BF16(a0.v, b.v, h0);           // B reused across both M tiles
            h1 = WMMA_BF16(a1.v, b.v, h1);
        }
        // C layout: VGPR r -> M = r (+8 upper lane half), N = lanelo
        {
            const int ncol = nt1 * 16 + lanelo;
            const int mr0  = (mta * 16 + lhalf * 8) * HS_STRIDE + ncol;
            #pragma unroll
            for (int r = 0; r < 8; ++r) {
                Hs[mr0 + r * HS_STRIDE]                    = f32_to_bf16(gelu_tanh(h0[r]));
                Hs[mr0 + (16 + r) * HS_STRIDE]             = f32_to_bf16(gelu_tanh(h1[r]));
            }
        }
        __syncthreads();

        // -------- phase 2 --------
        ABfrag afr[4][2];                            // A fragments, K=64 chunk
        #pragma unroll
        for (int mt = 0; mt < 4; ++mt) {
            const int base = (mt * 16 + lanelo) * HS_STRIDE + lhalf * 8;
            #pragma unroll
            for (int kk = 0; kk < 2; ++kk) {
                afr[mt][kk].q[0] = *(const uint4*)(&Hs[base + kk * 32]);
                afr[mt][kk].q[1] = *(const uint4*)(&Hs[base + kk * 32 + 16]);
            }
        }
        #pragma unroll
        for (int j = 0; j < 8; ++j) {
            const unsigned short* bp2 =
                w2t + ((wave * 8 + j) * 16 + lanelo) * INTER + ci + lhalf * 8;
            #pragma unroll
            for (int kk = 0; kk < 2; ++kk) {
                ABfrag b;
                b.q[0] = *(const uint4*)(bp2 + kk * 32);
                b.q[1] = *(const uint4*)(bp2 + kk * 32 + 16);
                #pragma unroll
                for (int mt = 0; mt < 4; ++mt)       // B reused across 4 M tiles
                    acc[mt][j] = WMMA_BF16(afr[mt][kk].v, b.v, acc[mt][j]);
            }
        }
        __syncthreads();
    }

    // ---- epilogue: scale by summed routing weights, store fp32 ----
    const int colb = wave * 128 + lanelo;            // wave*8 N tiles * 16
    #pragma unroll
    for (int mt = 0; mt < 4; ++mt) {
        #pragma unroll
        for (int r = 0; r < 8; ++r) {
            const int row = rowbase + mt * 16 + lhalf * 8 + r;
            const float ws = wsum[row];
            float* op = out + row * HID + colb;
            #pragma unroll
            for (int j = 0; j < 8; ++j)
                op[j * 16] = acc[mt][j][r] * ws;
        }
    }
}

// ---------------------------- launcher -------------------------------------

extern "C" void kernel_launch(void* const* d_in, const int* in_sizes, int n_in,
                              void* d_out, int out_size, void* d_ws, size_t ws_size,
                              hipStream_t stream) {
    (void)in_sizes; (void)n_in; (void)out_size; (void)ws_size;
    const float* x  = (const float*)d_in[0];
    // d_in[1] = scores      (unused: routing permutation cancels in combine)
    const float* ew = (const float*)d_in[2];
    // d_in[3] = top_experts (unused: all experts share w1/w2)
    const float* w1 = (const float*)d_in[4];
    const float* w2 = (const float*)d_in[5];
    float* out = (float*)d_out;

    char* ws = (char*)d_ws;
    const size_t XB_BYTES  = (size_t)T_TOKENS * HID * 2;   //  8 MB
    const size_t W1T_BYTES = (size_t)INTER * HID * 2;      // 16 MB
    const size_t W2T_BYTES = (size_t)HID * INTER * 2;      // 16 MB
    unsigned short* xb   = (unsigned short*)ws;
    unsigned short* w1t  = (unsigned short*)(ws + XB_BYTES);
    unsigned short* w2t  = (unsigned short*)(ws + XB_BYTES + W1T_BYTES);
    float*          wsum = (float*)(ws + XB_BYTES + W1T_BYTES + W2T_BYTES);

    k_cvt_x<<<(T_TOKENS * HID) / 256, 256, 0, stream>>>(x, xb);
    k_wsum <<<(T_TOKENS + 255) / 256, 256, 0, stream>>>(ew, wsum);
    k_tr_w1<<<(INTER * HID) / 256, 256, 0, stream>>>(w1, w1t);
    k_tr_w2<<<(HID * INTER) / 256, 256, 0, stream>>>(w2, w2t);

    k_moe_mlp<<<T_TOKENS / BM, 256, 0, stream>>>(xb, w1t, w2t, wsum, out);
}
